// TransformerEncoderBlock_50457275793599
// MI455X (gfx1250) — matrix operand order repaired and checked by reading, not compiled, not run on hardware
//
#include <hip/hip_runtime.h>
#include <hip/hip_bf16.h>
#include <math.h>

// ---------------------------------------------------------------------------
// Transformer encoder block for gfx1250 (MI455X), wave32 + WMMA + TDM.
//   h   = LN(x);  q,k,v = heads(h@W + b)           [bf16 WMMA 16x16x32]
//   E   = q k^T / 8  -> written straight into d_out attention region (fp32)
//   ctx = E @ v (reference bug preserved)           [bf16 WMMA, fp32 E cvt'd]
//   att = softmax(E) in place in d_out
//   x1  = x + ctx@wo + bo;  out = x1 + gelu(LN(x1)@w1+b1)@w2 + b2
// Main GEMM stages its A tile through LDS with the Tensor Data Mover
// (tensor_load_to_lds + s_wait_tensorcnt), double buffered.
// ---------------------------------------------------------------------------

typedef __attribute__((ext_vector_type(16))) __bf16 v16bf;
typedef __attribute__((ext_vector_type(8)))  float  v8f;

#define DEV static __device__ __forceinline__

DEV unsigned short f2bf(float f) {              // round-to-nearest-even fp32->bf16
  unsigned int u = __builtin_bit_cast(unsigned int, f);
  unsigned int r = u + 0x7fffu + ((u >> 16) & 1u);
  return (unsigned short)(r >> 16);
}

union Frag { unsigned int u[8]; v16bf v; };     // 8 dwords = 16 bf16 per lane

// ---------------------------------------------------------------------------
// TDM support (probe-verified builtin; arity differs between toolchains).
// ---------------------------------------------------------------------------
#if defined(__HIP_DEVICE_COMPILE__) && defined(__has_builtin)
#  if __has_builtin(__builtin_amdgcn_tensor_load_to_lds) && \
      __has_builtin(__builtin_amdgcn_s_wait_tensorcnt)
#    define USE_TDM 1
#  endif
#endif
#ifndef USE_TDM
#  define USE_TDM 0
#endif

#if USE_TDM
typedef __attribute__((ext_vector_type(4))) unsigned int u32x4;
typedef __attribute__((ext_vector_type(8))) int i32x8;
typedef __attribute__((ext_vector_type(4))) int i32x4;

// 2D tile load: bf16 tensor [tensor_h][tensor_w] row-major, tile tile_h x tile_w
// starting at gptr (tile origin), into LDS at byte offset lds_off.
// D# layout per CDNA5 ISA ch.8 (group0: count/type/addrs, group1: dims/strides).
DEV void tdm_load_tile_2d(unsigned lds_off, const void* gptr,
                          int tensor_w, int tensor_h, int tile_w, int tile_h) {
  unsigned long long ga = (unsigned long long)(size_t)gptr;
  u32x4 g0;
  g0[0] = 1u;                                             // count=1, user D#
  g0[1] = lds_off;                                        // lds_addr (bytes)
  g0[2] = (unsigned)(ga & 0xffffffffull);                 // global_addr[31:0]
  g0[3] = (unsigned)((ga >> 32) & 0x01ffffffull) | 0x80000000u; // [56:32]|type=2
  i32x8 g1;
  g1[0] = (int)(1u << 16);                                // data_size=1 (2 bytes)
  g1[1] = (int)(((unsigned)tensor_w & 0xffffu) << 16);    // tensor_dim0 lo16
  g1[2] = (int)((((unsigned)tensor_w) >> 16) |
                (((unsigned)tensor_h & 0xffffu) << 16));  // dim0 hi | dim1 lo
  g1[3] = (int)((((unsigned)tensor_h) >> 16) |
                (((unsigned)tile_w) << 16));              // dim1 hi | tile_dim0
  g1[4] = (int)(unsigned)tile_h;                          // tile_dim1 (dim2=0)
  g1[5] = tensor_w;                                       // dim0_stride lo32
  g1[6] = 0;                                              // stride hi bits
  g1[7] = 0;
  i32x4 z4 = {0, 0, 0, 0};
#if defined(__clang_major__) && __clang_major__ >= 23
  i32x8 z8 = {0, 0, 0, 0, 0, 0, 0, 0};
  __builtin_amdgcn_tensor_load_to_lds(g0, g1, z4, z4, z8, 0);
#else
  __builtin_amdgcn_tensor_load_to_lds(g0, g1, z4, z4, 0);
#endif
}
#endif  // USE_TDM

// ---------------------------------------------------------------------------
// Weight convert + transpose: wt[n*K + k] = bf16(w[k*N + n])
// ---------------------------------------------------------------------------
__global__ __launch_bounds__(256)
void convert_transpose_kernel(const float* __restrict__ w,
                              unsigned short* __restrict__ wt,
                              int K, int N) {
  size_t idx = (size_t)blockIdx.x * 256 + threadIdx.x;
  size_t total = (size_t)K * N;
  if (idx < total) {
    int k = (int)(idx / N);
    int n = (int)(idx % N);
    wt[(size_t)n * K + k] = f2bf(w[idx]);
  }
}

// ---------------------------------------------------------------------------
// LayerNorm over D=768, one row per block, bf16 output for WMMA consumption.
// ---------------------------------------------------------------------------
__global__ __launch_bounds__(256)
void layernorm_bf16_kernel(const float* __restrict__ x,
                           const float* __restrict__ g,
                           const float* __restrict__ be,
                           unsigned short* __restrict__ out) {
  __shared__ float sred[256];
  const int tid = threadIdx.x;
  const size_t roff = (size_t)blockIdx.x * 768;
  float v0 = x[roff + tid];
  float v1 = x[roff + tid + 256];
  float v2 = x[roff + tid + 512];
  sred[tid] = v0 + v1 + v2;
  __syncthreads();
  for (int s = 128; s > 0; s >>= 1) {
    if (tid < s) sred[tid] += sred[tid + s];
    __syncthreads();
  }
  const float mu = sred[0] * (1.0f / 768.0f);
  __syncthreads();
  const float d0 = v0 - mu, d1 = v1 - mu, d2 = v2 - mu;
  sred[tid] = d0 * d0 + d1 * d1 + d2 * d2;
  __syncthreads();
  for (int s = 128; s > 0; s >>= 1) {
    if (tid < s) sred[tid] += sred[tid + s];
    __syncthreads();
  }
  const float r = rsqrtf(sred[0] * (1.0f / 768.0f) + 1e-5f);
  out[roff + tid      ] = f2bf(d0 * r * g[tid      ] + be[tid      ]);
  out[roff + tid + 256] = f2bf(d1 * r * g[tid + 256] + be[tid + 256]);
  out[roff + tid + 512] = f2bf(d2 * r * g[tid + 512] + be[tid + 512]);
}

// ---------------------------------------------------------------------------
// In-place softmax over rows of 2048 (operates on d_out attention region).
// ---------------------------------------------------------------------------
__global__ __launch_bounds__(256)
void softmax_inplace_kernel(float* __restrict__ att) {
  __shared__ float sred[256];
  const int tid = threadIdx.x;
  float* p = att + (size_t)blockIdx.x * 2048;
  float v[8];
  float mx = -3.402823466e+38f;
#pragma unroll
  for (int i = 0; i < 8; ++i) { v[i] = p[tid + 256 * i]; mx = fmaxf(mx, v[i]); }
  sred[tid] = mx;
  __syncthreads();
  for (int s = 128; s > 0; s >>= 1) {
    if (tid < s) sred[tid] = fmaxf(sred[tid], sred[tid + s]);
    __syncthreads();
  }
  mx = sred[0];
  __syncthreads();
  float sum = 0.0f;
#pragma unroll
  for (int i = 0; i < 8; ++i) { v[i] = expf(v[i] - mx); sum += v[i]; }
  sred[tid] = sum;
  __syncthreads();
  for (int s = 128; s > 0; s >>= 1) {
    if (tid < s) sred[tid] += sred[tid + s];
    __syncthreads();
  }
  const float inv = 1.0f / sred[0];
#pragma unroll
  for (int i = 0; i < 8; ++i) p[tid + 256 * i] = v[i] * inv;
}

// ---------------------------------------------------------------------------
// bf16 -> bf16 transpose of v heads: [BH][2048][64] -> [BH][64][2048]
// 32x32 LDS tile, padded to kill bank conflicts.
// ---------------------------------------------------------------------------
__global__ __launch_bounds__(256)
void transpose_v_kernel(const unsigned short* __restrict__ vb,
                        unsigned short* __restrict__ vt) {
  __shared__ unsigned short tile[32][33];
  const int bh = blockIdx.z;
  const int s0 = blockIdx.y * 32;
  const int d0 = blockIdx.x * 32;
  const int tx = threadIdx.x & 31, ty = threadIdx.x >> 5;
  const unsigned short* src = vb + (size_t)bh * 2048 * 64;
  unsigned short* dst = vt + (size_t)bh * 64 * 2048;
#pragma unroll
  for (int i = 0; i < 4; ++i)
    tile[ty + 8 * i][tx] = src[(size_t)(s0 + ty + 8 * i) * 64 + d0 + tx];
  __syncthreads();
#pragma unroll
  for (int i = 0; i < 4; ++i)
    dst[(size_t)(d0 + ty + 8 * i) * 2048 + s0 + tx] = tile[tx][ty + 8 * i];
}

// ---------------------------------------------------------------------------
// bf16 WMMA GEMM: C[M,N] = A[M,K] @ B[K,N] (+bias, epilogue per MODE)
//   A: bf16 [M,K] row-major.   Bt: bf16 [N,K] (i.e. B transposed).
// Block = 256 thr = 8 waves, tile 64(M) x 128(N); wave tile 32x32 (2x2 frags).
// A tile (64x32 bf16) is staged through LDS by the Tensor Data Mover,
// double buffered; B fragments load straight from global (contiguous pairs).
// Fragment layouts per CDNA5 ISA 7.12.2 (wave32):
//   A dword j (0..7): row = lane&15, k = (j>>2)*16 + (lane>>4)*8 + 2*(j&3)
//   B dword j (0..7): col = lane&15, k = (j>>2)*16 + (lane>>4)*8 + 2*(j&3)
//   C dword j (0..7): row = j + (lane>>4)*8, col = lane&15
// ---------------------------------------------------------------------------
#define MODE_HEADS_BF16 0   // out bf16 in [B,H,S,hd] heads layout
#define MODE_F32_RESID  1   // out fp32 row-major, + residual
#define MODE_BF16_GELU  2   // out bf16 row-major, exact gelu

template <int MODE>
__global__ __launch_bounds__(256)
void gemm_bf16_kernel(const unsigned short* __restrict__ A,
                      const unsigned short* __restrict__ Bt,
                      const float* __restrict__ bias,
                      const float* __restrict__ resid,
                      float* __restrict__ outF,
                      unsigned short* __restrict__ outB,
                      int M, int N, int K) {
  const int tid  = threadIdx.x;
  const int wave = tid >> 5;
  const int lane = tid & 31;
  const int row  = lane & 15;
  const int hi   = lane >> 4;
  const int mblk  = blockIdx.y * 64;
  const int mbase = mblk + (wave & 1) * 32;
  const int nbase = blockIdx.x * 128 + (wave >> 1) * 32;
  const int nIter = K >> 5;

  v8f acc[2][2];
#pragma unroll
  for (int mt = 0; mt < 2; ++mt)
#pragma unroll
    for (int nt = 0; nt < 2; ++nt) acc[mt][nt] = (v8f)(0.0f);

#if USE_TDM
  __shared__ unsigned short atile[2][64 * 32];
  const int wid = __builtin_amdgcn_readfirstlane((int)threadIdx.x) >> 5;
  if (wid == 0) {
    tdm_load_tile_2d((unsigned)(size_t)(const void*)&atile[0][0],
                     A + (size_t)mblk * K, K, M, 32, 64);
  }
#endif

  for (int t = 0; t < nIter; ++t) {
    const int kk = t << 5;
    Frag a[2], b[2];

#if USE_TDM
    if (wid == 0) {
      if (t + 1 < nIter) {
        tdm_load_tile_2d((unsigned)(size_t)(const void*)&atile[(t + 1) & 1][0],
                         A + (size_t)mblk * K + (kk + 32), K, M, 32, 64);
        __builtin_amdgcn_s_wait_tensorcnt((short)1);
      } else {
        __builtin_amdgcn_s_wait_tensorcnt((short)0);
      }
    }
    __syncthreads();
    const unsigned int* asrc = (const unsigned int*)&atile[t & 1][0];
#pragma unroll
    for (int mt = 0; mt < 2; ++mt) {
      const int rl16 = ((wave & 1) * 32 + mt * 16 + row) * 16;  // dwords/row=16
#pragma unroll
      for (int j = 0; j < 8; ++j) {
        const int kd = (j >> 2) * 8 + hi * 4 + (j & 3);         // k0>>1
        a[mt].u[j] = asrc[rl16 + kd];
      }
    }
#else
#pragma unroll
    for (int mt = 0; mt < 2; ++mt) {
      const unsigned int* ap =
          (const unsigned int*)(A + (size_t)(mbase + mt * 16 + row) * K + kk);
#pragma unroll
      for (int j = 0; j < 8; ++j) {
        const int k0 = (j >> 2) * 16 + hi * 8 + 2 * (j & 3);
        a[mt].u[j] = ap[k0 >> 1];
      }
    }
#endif

#pragma unroll
    for (int nt = 0; nt < 2; ++nt) {
      const unsigned int* bp =
          (const unsigned int*)(Bt + (size_t)(nbase + nt * 16 + row) * K + kk);
      if (kk + 32 < K) __builtin_prefetch(bp + 16, 0, 0);  // global_prefetch_b8
#pragma unroll
      for (int j = 0; j < 8; ++j) b[nt].u[j] = bp[(j >> 2) * 8 + hi * 4 + (j & 3)];  // k0=(j>>2)*16+hi*8+2(j&3)
    }

#pragma unroll
    for (int mt = 0; mt < 2; ++mt)
#pragma unroll
      for (int nt = 0; nt < 2; ++nt)
        acc[mt][nt] = __builtin_amdgcn_wmma_f32_16x16x32_bf16(
            false, a[mt].v, false, b[nt].v, (short)0, acc[mt][nt], false, false);

#if USE_TDM
    __syncthreads();   // all waves done reading this slot before TDM reuses it
#endif
  }

#pragma unroll
  for (int mt = 0; mt < 2; ++mt) {
#pragma unroll
    for (int nt = 0; nt < 2; ++nt) {
#pragma unroll
      for (int j = 0; j < 8; ++j) {
        const int m = mbase + mt * 16 + j + hi * 8;
        const int n = nbase + nt * 16 + row;
        float val = acc[mt][nt][j] + bias[n];
        if (MODE == MODE_F32_RESID) {
          outF[(size_t)m * N + n] = val + resid[(size_t)m * N + n];
        } else if (MODE == MODE_BF16_GELU) {
          val = 0.5f * val * (1.0f + erff(val * 0.70710678118654752f));
          outB[(size_t)m * N + n] = f2bf(val);
        } else {  // MODE_HEADS_BF16: m=(b,s), n=(h,d) -> [B,H,S,hd]
          const int bb = m >> 11, ss = m & 2047;
          const int hh = n >> 6,  dd = n & 63;
          outB[(((size_t)(bb * 12 + hh) * 2048 + ss) << 6) + dd] = f2bf(val);
        }
      }
    }
  }
}

// ---------------------------------------------------------------------------
// energy[bh, m, n] = (q[bh,m,:] . k[bh,n,:]) / 8    (bf16 WMMA, fp32 out)
// q,k: bf16 [BH, 2048, 64].  Out rows of 2048 fp32 in d_out attention region.
// ---------------------------------------------------------------------------
__global__ __launch_bounds__(256)
void energy_kernel(const unsigned short* __restrict__ q,
                   const unsigned short* __restrict__ k,
                   float* __restrict__ eout) {
  const int tid  = threadIdx.x;
  const int wave = tid >> 5;
  const int lane = tid & 31;
  const int row  = lane & 15;
  const int hi   = lane >> 4;
  const int bh   = blockIdx.z;
  const unsigned short* Aq = q + (size_t)bh * 2048 * 64;
  const unsigned short* Bk = k + (size_t)bh * 2048 * 64;  // B^T == k rows
  float* out = eout + (size_t)bh * 2048 * 2048;

  const int mbase = blockIdx.y * 64  + (wave & 1) * 32;
  const int nbase = blockIdx.x * 128 + (wave >> 1) * 32;

  v8f acc[2][2];
#pragma unroll
  for (int mt = 0; mt < 2; ++mt)
#pragma unroll
    for (int nt = 0; nt < 2; ++nt) acc[mt][nt] = (v8f)(0.0f);

#pragma unroll
  for (int kk = 0; kk < 64; kk += 32) {
    Frag a[2], b[2];
#pragma unroll
    for (int mt = 0; mt < 2; ++mt) {
      const unsigned int* ap =
          (const unsigned int*)(Aq + (size_t)(mbase + mt * 16 + row) * 64 + kk);
#pragma unroll
      for (int j = 0; j < 8; ++j) {
        const int k0 = (j >> 2) * 16 + hi * 8 + 2 * (j & 3);
        a[mt].u[j] = ap[k0 >> 1];
      }
    }
#pragma unroll
    for (int nt = 0; nt < 2; ++nt) {
      const unsigned int* bp =
          (const unsigned int*)(Bk + (size_t)(nbase + nt * 16 + row) * 64 + kk);
#pragma unroll
      for (int j = 0; j < 8; ++j) b[nt].u[j] = bp[(j >> 2) * 8 + hi * 4 + (j & 3)];
    }
#pragma unroll
    for (int mt = 0; mt < 2; ++mt)
#pragma unroll
      for (int nt = 0; nt < 2; ++nt)
        acc[mt][nt] = __builtin_amdgcn_wmma_f32_16x16x32_bf16(
            false, a[mt].v, false, b[nt].v, (short)0, acc[mt][nt], false, false);
  }

#pragma unroll
  for (int mt = 0; mt < 2; ++mt)
#pragma unroll
    for (int nt = 0; nt < 2; ++nt)
#pragma unroll
      for (int j = 0; j < 8; ++j) {
        const int m = mbase + mt * 16 + j + hi * 8;
        const int n = nbase + nt * 16 + row;
        out[(size_t)m * 2048 + n] = acc[mt][nt][j] * 0.125f;
      }
}

// ---------------------------------------------------------------------------
// ctx[bh, m, d] = sum_n energy[bh,m,n] * v[bh,n,d]   (bf16 WMMA 16x16x32)
// Raw (pre-softmax) fp32 energy is converted to bf16 fragments in-register
// (fp32 accumulation stays in the WMMA). vt is v transposed: [BH][64][2048].
// Output merged heads bf16 [B,S,D] for the wo projection.
// ---------------------------------------------------------------------------
__global__ __launch_bounds__(256)
void ctx_bf16_kernel(const float* __restrict__ energy,
                     const unsigned short* __restrict__ vt,
                     unsigned short* __restrict__ ctxb) {
  const int tid  = threadIdx.x;
  const int wave = tid >> 5;
  const int lane = tid & 31;
  const int row  = lane & 15;
  const int hi   = lane >> 4;
  const int bh = blockIdx.y;
  const int bb = bh / 12, hh = bh % 12;
  const float* E = energy + (size_t)bh * 2048 * 2048;
  const unsigned short* Vt = vt + (size_t)bh * 64 * 2048;

  const int mbase = blockIdx.x * 128 + wave * 16;

  v8f acc[4];
#pragma unroll
  for (int nt = 0; nt < 4; ++nt) acc[nt] = (v8f)(0.0f);

  const float* ar = E + (size_t)(mbase + row) * 2048;
  for (int kk = 0; kk < 2048; kk += 32) {
    float g0v[8], g1v[8];
#pragma unroll
    for (int i = 0; i < 8; ++i) g0v[i] = ar[kk + hi * 8 + i];
#pragma unroll
    for (int i = 0; i < 8; ++i) g1v[i] = ar[kk + 16 + hi * 8 + i];
    Frag a;
#pragma unroll
    for (int jj = 0; jj < 4; ++jj) {
      a.u[jj]     = (unsigned)f2bf(g0v[2 * jj]) |
                    ((unsigned)f2bf(g0v[2 * jj + 1]) << 16);
      a.u[4 + jj] = (unsigned)f2bf(g1v[2 * jj]) |
                    ((unsigned)f2bf(g1v[2 * jj + 1]) << 16);
    }
#pragma unroll
    for (int nt = 0; nt < 4; ++nt) {
      const unsigned int* bp =
          (const unsigned int*)(Vt + (size_t)(nt * 16 + row) * 2048 + kk);
      Frag b;
#pragma unroll
      for (int j = 0; j < 8; ++j) b.u[j] = bp[(j >> 2) * 8 + hi * 4 + (j & 3)];
      acc[nt] = __builtin_amdgcn_wmma_f32_16x16x32_bf16(
          false, a.v, false, b.v, (short)0, acc[nt], false, false);
    }
  }

#pragma unroll
  for (int nt = 0; nt < 4; ++nt)
#pragma unroll
    for (int j = 0; j < 8; ++j) {
      const int m = mbase + j + hi * 8;
      const int n = nt * 16 + row;
      ctxb[((size_t)(bb * 2048 + m)) * 768 + hh * 64 + n] = f2bf(acc[nt][j]);
    }
}

// ---------------------------------------------------------------------------
// Host-side orchestration.
// ---------------------------------------------------------------------------
extern "C" void kernel_launch(void* const* d_in, const int* in_sizes, int n_in,
                              void* d_out, int out_size, void* d_ws, size_t ws_size,
                              hipStream_t stream) {
  (void)in_sizes; (void)n_in; (void)out_size; (void)ws_size;
  constexpr int S = 2048, D = 768, F = 3072, BH = 24, M = 4096;  // B*S = 4096

  const float* x  = (const float*)d_in[0];
  const float* wq = (const float*)d_in[1];
  const float* bq = (const float*)d_in[2];
  const float* wk = (const float*)d_in[3];
  const float* bk = (const float*)d_in[4];
  const float* wv = (const float*)d_in[5];
  const float* bv = (const float*)d_in[6];
  const float* wo = (const float*)d_in[7];
  const float* bo = (const float*)d_in[8];
  const float* w1 = (const float*)d_in[9];
  const float* b1 = (const float*)d_in[10];
  const float* w2 = (const float*)d_in[11];
  const float* b2 = (const float*)d_in[12];
  const float* g1 = (const float*)d_in[13];
  const float* be1 = (const float*)d_in[14];
  const float* g2 = (const float*)d_in[15];
  const float* be2 = (const float*)d_in[16];

  char* ws = (char*)d_ws;
  size_t off = 0;
  auto alloc = [&](size_t bytes) -> void* {
    void* p = ws + off;
    off += (bytes + 255) & ~(size_t)255;
    return p;
  };
  unsigned short* wtq  = (unsigned short*)alloc((size_t)D * D * 2);
  unsigned short* wtk  = (unsigned short*)alloc((size_t)D * D * 2);
  unsigned short* wtv  = (unsigned short*)alloc((size_t)D * D * 2);
  unsigned short* wto  = (unsigned short*)alloc((size_t)D * D * 2);
  unsigned short* wt1  = (unsigned short*)alloc((size_t)D * F * 2);
  unsigned short* wt2  = (unsigned short*)alloc((size_t)F * D * 2);
  unsigned short* hb   = (unsigned short*)alloc((size_t)M * D * 2);
  unsigned short* qb   = (unsigned short*)alloc((size_t)BH * S * 64 * 2);
  unsigned short* kb   = (unsigned short*)alloc((size_t)BH * S * 64 * 2);
  unsigned short* vb   = (unsigned short*)alloc((size_t)BH * S * 64 * 2);
  unsigned short* vtb  = (unsigned short*)alloc((size_t)BH * 64 * S * 2);
  unsigned short* ctxb = (unsigned short*)alloc((size_t)M * D * 2);
  float*          x1f  = (float*)alloc((size_t)M * D * 4);
  unsigned short* h2b  = (unsigned short*)alloc((size_t)M * D * 2);
  unsigned short* mlph = (unsigned short*)alloc((size_t)M * F * 2);

  float* xout = (float*)d_out;                       // [2,2048,768]
  float* att  = xout + (size_t)M * D;                // [2,12,2048,2048]

  const dim3 blk(256);

  // Weight conversion (fp32 -> bf16, transposed to [N][K]).
  convert_transpose_kernel<<<dim3((D * D + 255) / 256), blk, 0, stream>>>(wq, wtq, D, D);
  convert_transpose_kernel<<<dim3((D * D + 255) / 256), blk, 0, stream>>>(wk, wtk, D, D);
  convert_transpose_kernel<<<dim3((D * D + 255) / 256), blk, 0, stream>>>(wv, wtv, D, D);
  convert_transpose_kernel<<<dim3((D * D + 255) / 256), blk, 0, stream>>>(wo, wto, D, D);
  convert_transpose_kernel<<<dim3((D * F + 255) / 256), blk, 0, stream>>>(w1, wt1, D, F);
  convert_transpose_kernel<<<dim3((F * D + 255) / 256), blk, 0, stream>>>(w2, wt2, F, D);

  // h = LN(x)
  layernorm_bf16_kernel<<<dim3(M), blk, 0, stream>>>(x, g1, be1, hb);

  // q, k, v projections (heads layout bf16)
  const dim3 gproj(D / 128, M / 64);
  gemm_bf16_kernel<MODE_HEADS_BF16><<<gproj, blk, 0, stream>>>(hb, wtq, bq, nullptr, nullptr, qb, M, D, D);
  gemm_bf16_kernel<MODE_HEADS_BF16><<<gproj, blk, 0, stream>>>(hb, wtk, bk, nullptr, nullptr, kb, M, D, D);
  gemm_bf16_kernel<MODE_HEADS_BF16><<<gproj, blk, 0, stream>>>(hb, wtv, bv, nullptr, nullptr, vb, M, D, D);

  // v transposed per head: [BH][2048][64] -> [BH][64][2048]
  transpose_v_kernel<<<dim3(2, S / 32, BH), blk, 0, stream>>>(vb, vtb);

  // energy -> d_out attention region (raw, pre-softmax)
  energy_kernel<<<dim3(S / 128, S / 64, BH), blk, 0, stream>>>(qb, kb, att);

  // ctx = energy @ v  (reads raw energy BEFORE in-place softmax)
  ctx_bf16_kernel<<<dim3(S / 128, BH), blk, 0, stream>>>(att, vtb, ctxb);

  // attention = softmax(energy), in place in d_out
  softmax_inplace_kernel<<<dim3(BH * S), blk, 0, stream>>>(att);

  // x1 = x + ctx @ wo + bo
  gemm_bf16_kernel<MODE_F32_RESID><<<gproj, blk, 0, stream>>>(ctxb, wto, bo, x, x1f, nullptr, M, D, D);

  // h2 = LN(x1)
  layernorm_bf16_kernel<<<dim3(M), blk, 0, stream>>>(x1f, g2, be2, h2b);

  // mlp hidden = gelu(h2 @ w1 + b1)
  gemm_bf16_kernel<MODE_BF16_GELU><<<dim3(F / 128, M / 64), blk, 0, stream>>>(h2b, wt1, b1, nullptr, nullptr, mlph, M, F, D);

  // out x = x1 + hidden @ w2 + b2
  gemm_bf16_kernel<MODE_F32_RESID><<<gproj, blk, 0, stream>>>(mlph, wt2, b2, x1f, xout, nullptr, M, D, F);
}
